// ScaledDotProductAttention_10376640987807
// MI455X (gfx1250) — matrix operand order repaired and checked by reading, not compiled, not run on hardware
//
#include <hip/hip_runtime.h>

typedef __attribute__((ext_vector_type(16))) _Float16 v16h;
typedef __attribute__((ext_vector_type(8)))  float    v8f;

#define WAVES 8
#define TPW   16          // key tiles (16 wide) per wave -> 256 cols/wave
#define NSEQ  2048
#define DHEAD 64
#define QTILE 16
#define CHS   34          // halfs per row in chunk LDS (pad 32->34)

__global__ __launch_bounds__(256)
void attn_fused_kernel(const float* __restrict__ q, const float* __restrict__ k,
                       const float* __restrict__ v, const float* __restrict__ mask,
                       float* __restrict__ out, float* __restrict__ attn)
{
    __shared__ _Float16 chunkLds[WAVES][QTILE * CHS];  // per-wave prob transpose bounce
    __shared__ float    redbuf[WAVES][QTILE];          // cross-wave softmax partials
    __shared__ float    gvals[QTILE];                  // row max, then 1/rowsum
    __shared__ float    outred[QTILE * DHEAD];         // cross-wave output reduce

    const int tid  = threadIdx.x;
    const int wave = tid >> 5;
    const int lane = tid & 31;
    const int l16  = lane & 15;
    const int hi   = (lane >> 4) & 1;   // lane half: 0 -> lanes 0-15, 1 -> 16-31

    const int bh    = blockIdx.x >> 7;   // 32 (b,h) heads
    const int qbase = (blockIdx.x & 127) * QTILE;

    const size_t headOff = (size_t)bh * NSEQ * DHEAD;
    const float* qh = q    + headOff;
    const float* kh = k    + headOff;
    const float* vh = v    + headOff;
    const float* mh = mask + headOff;

    for (int i = tid; i < QTILE * DHEAD; i += 256) outred[i] = 0.0f;

    //------------------------------------------------------------------
    // A fragments for GEMM1: qm = (q + mask) / 8, 16 rows x 64 (two K=32 chunks)
    // f16 A layout: row m = lane&15; lo lanes K = 0..7,16..23; hi lanes +8
    //------------------------------------------------------------------
    v16h a0, a1;
    {
        const int kb = hi ? 8 : 0;
        const float* qrow = qh + (size_t)(qbase + l16) * DHEAD;
        const float* mrow = mh + (size_t)(qbase + l16) * DHEAD;
        #pragma unroll
        for (int c = 0; c < 2; ++c) {
            #pragma unroll
            for (int j = 0; j < 8; ++j) {
                const int d = 32 * c + kb + 2 * j + ((j >= 4) ? 8 : 0);
                float2 qv = *(const float2*)(qrow + d);
                float2 mv = *(const float2*)(mrow + d);
                _Float16 h0 = (_Float16)((qv.x + mv.x) * 0.125f);
                _Float16 h1 = (_Float16)((qv.y + mv.y) * 0.125f);
                if (c == 0) { a0[2*j] = h0; a0[2*j+1] = h1; }
                else        { a1[2*j] = h0; a1[2*j+1] = h1; }
            }
        }
    }

    //------------------------------------------------------------------
    // GEMM1: S = qm @ k^T ; wave owns 16x256 strip in registers
    // f16 B layout: col n = lane&15; lo lanes K = 0..7,16..23; hi lanes +8
    //------------------------------------------------------------------
    v8f acc[TPW];
    {
        const int kb2 = hi ? 8 : 0;
        #pragma unroll
        for (int t = 0; t < TPW; ++t) {
            const int keyCol = (wave * TPW + t) * 16 + l16;
            const float* krow = kh + (size_t)keyCol * DHEAD;
            v8f c = {};
            #pragma unroll
            for (int cc = 0; cc < 2; ++cc) {
                v16h b;
                #pragma unroll
                for (int j = 0; j < 8; ++j) {
                    float2 kv = *(const float2*)(krow + 32 * cc + kb2 + 2 * j + ((j >= 4) ? 8 : 0));
                    b[2*j]   = (_Float16)kv.x;
                    b[2*j+1] = (_Float16)kv.y;
                }
                c = __builtin_amdgcn_wmma_f32_16x16x32_f16(
                        false, (cc == 0 ? a0 : a1), false, b, (short)0, c, false, false);
            }
            acc[t] = c;
        }
    }

    //------------------------------------------------------------------
    // Softmax: C/D layout -> row = r + hi*8, col = tile*16 + (lane&15)
    //------------------------------------------------------------------
    float red8[8];
    #pragma unroll
    for (int r = 0; r < 8; ++r) {
        float mx = acc[0][r];
        #pragma unroll
        for (int t = 1; t < TPW; ++t) mx = fmaxf(mx, acc[t][r]);
        #pragma unroll
        for (int s = 1; s <= 8; s <<= 1) mx = fmaxf(mx, __shfl_xor(mx, s, 32));
        red8[r] = mx;
    }
    if (l16 == 0) {
        #pragma unroll
        for (int r = 0; r < 8; ++r) redbuf[wave][hi * 8 + r] = red8[r];
    }
    __syncthreads();
    if (tid < QTILE) {
        float mx = redbuf[0][tid];
        for (int w2 = 1; w2 < WAVES; ++w2) mx = fmaxf(mx, redbuf[w2][tid]);
        gvals[tid] = mx;
    }
    __syncthreads();

    #pragma unroll
    for (int r = 0; r < 8; ++r) {
        const float rm = gvals[hi * 8 + r];
        float sm = 0.0f;
        #pragma unroll
        for (int t = 0; t < TPW; ++t) {
            float p = __expf(acc[t][r] - rm);
            acc[t][r] = p;
            sm += p;
        }
        #pragma unroll
        for (int s = 1; s <= 8; s <<= 1) sm += __shfl_xor(sm, s, 32);
        red8[r] = sm;
    }
    if (l16 == 0) {
        #pragma unroll
        for (int r = 0; r < 8; ++r) redbuf[wave][hi * 8 + r] = red8[r];
    }
    __syncthreads();
    if (tid < QTILE) {
        float s = 0.0f;
        for (int w2 = 0; w2 < WAVES; ++w2) s += redbuf[w2][tid];
        gvals[tid] = 1.0f / s;
    }
    __syncthreads();

    // Scale in registers and stream attn to HBM (write-once -> non-temporal)
    #pragma unroll
    for (int r = 0; r < 8; ++r) {
        const int row = hi * 8 + r;
        const float inv = gvals[row];
        const size_t rowBase = ((size_t)bh * NSEQ + qbase + row) * (size_t)NSEQ;
        #pragma unroll
        for (int t = 0; t < TPW; ++t) {
            float val = acc[t][r] * inv;
            acc[t][r] = val;
            __builtin_nontemporal_store(
                val, attn + rowBase + (size_t)(wave * TPW + t) * 16 + l16);
        }
    }

    //------------------------------------------------------------------
    // GEMM2: out += attn_strip @ v ; transpose bounce via wave-private LDS
    //------------------------------------------------------------------
    v8f o[4];
    #pragma unroll
    for (int nt = 0; nt < 4; ++nt) o[nt] = (v8f){};

    _Float16* chunk = chunkLds[wave];
    {
        const int kb2 = hi ? 16 : 0;
        const int kb  = hi ? 8 : 0;
        #pragma unroll
        for (int s = 0; s < 8; ++s) {
            // write this wave's 16x32 prob chunk (tiles 2s, 2s+1) as f16
            #pragma unroll
            for (int r = 0; r < 8; ++r) {
                const int row = hi * 8 + r;
                chunk[row * CHS + l16]      = (_Float16)acc[2*s][r];
                chunk[row * CHS + 16 + l16] = (_Float16)acc[2*s+1][r];
            }
            // reload in f16 A layout (DS ops are in-order within a wave)
            v16h a;
            #pragma unroll
            for (int j = 0; j < 8; ++j) {
                const int kk = kb + 2 * j + ((j >= 4) ? 8 : 0);
                a[2*j]   = chunk[l16 * CHS + kk];
                a[2*j+1] = chunk[l16 * CHS + kk + 1];
            }
            const int kvbase = wave * 256 + 32 * s;
            #pragma unroll
            for (int nt = 0; nt < 4; ++nt) {
                const float* vcol = vh + (size_t)(kvbase + kb) * DHEAD + nt * 16 + l16;
                v16h b;
                #pragma unroll
                for (int j = 0; j < 8; ++j) {
                    b[2*j]   = (_Float16)vcol[(2*j + ((j >= 4) ? 8 : 0)) * DHEAD];
                    b[2*j+1] = (_Float16)vcol[(2*j + 1 + ((j >= 4) ? 8 : 0)) * DHEAD];
                }
                o[nt] = __builtin_amdgcn_wmma_f32_16x16x32_f16(
                            false, a, false, b, (short)0, o[nt], false, false);
            }
        }
    }

    // cross-wave reduction of 16x64 partial outputs
    #pragma unroll
    for (int nt = 0; nt < 4; ++nt) {
        #pragma unroll
        for (int r = 0; r < 8; ++r) {
            const int row = hi * 8 + r;
            const int col = nt * 16 + l16;
            atomicAdd(&outred[row * DHEAD + col], o[nt][r]);
        }
    }
    __syncthreads();

    {
        const int i   = tid * 4;           // 256 threads x 4 floats = 16x64
        const int row = i >> 6;
        const int col = i & 63;
        float4 val = *(const float4*)(outred + i);
        *(float4*)(out + ((size_t)bh * NSEQ + qbase + row) * DHEAD + col) = val;
    }
}

extern "C" void kernel_launch(void* const* d_in, const int* in_sizes, int n_in,
                              void* d_out, int out_size, void* d_ws, size_t ws_size,
                              hipStream_t stream) {
    (void)in_sizes; (void)n_in; (void)out_size; (void)d_ws; (void)ws_size;
    const float* q    = (const float*)d_in[0];
    const float* k    = (const float*)d_in[1];
    const float* v    = (const float*)d_in[2];
    const float* mask = (const float*)d_in[3];
    float* out  = (float*)d_out;
    float* attn = out + (size_t)4 * 8 * 2048 * 64;   // outputs concatenated: output, attn

    dim3 grid(32 * 128);   // 32 heads x 128 query tiles of 16 rows
    dim3 block(256);       // 8 wave32
    hipLaunchKernelGGL(attn_fused_kernel, grid, block, 0, stream,
                       q, k, v, mask, out, attn);
}
